// DecoderWithAttention_52072183496710
// MI455X (gfx1250) — matrix operand order repaired and checked by reading, not compiled, not run on hardware
//
#include <hip/hip_runtime.h>
#include <hip/hip_bf16.h>
#include <cstdint>
#include <cstddef>

// Problem dims
#define B_   128
#define LCAP 32
#define P_   196
#define D_   2048
#define A_   512
#define E_   512
#define H_   512
#define V_   10000
#define T_   31

typedef __attribute__((ext_vector_type(16))) __bf16 v16bf;
typedef __attribute__((ext_vector_type(8)))  float  v8f;

__device__ __forceinline__ float bf2f(uint16_t u) {
    union { uint32_t u; float f; } x; x.u = ((uint32_t)u) << 16; return x.f;
}
__device__ __forceinline__ uint16_t f2bf(float f) {
    union { float f; uint32_t u; } x; x.f = f;
    uint32_t r = (x.u + 0x7fffu + ((x.u >> 16) & 1u)) >> 16;
    return (uint16_t)r;
}
__device__ __forceinline__ float sigm(float x) { return 1.f / (1.f + __expf(-x)); }

union FragA { v16bf v; uint32_t u[8]; };
union FragB { v16bf v; uint4 q[2]; };

// ---------------------------------------------------------------------------
// Stable descending argsort of lengths (B=128), emit order/dec_len/caps outs.
// ---------------------------------------------------------------------------
__global__ void sort_kernel(const int* __restrict__ lengths,
                            const int* __restrict__ captions,
                            int* __restrict__ order, int* __restrict__ declen,
                            int* __restrict__ caps_s,
                            float* __restrict__ out_caps,
                            float* __restrict__ out_declen,
                            float* __restrict__ out_order) {
    __shared__ int len[B_];
    __shared__ int ord[B_];
    int b = threadIdx.x;
    len[b] = lengths[b];
    __syncthreads();
    int L0 = len[b];
    int rank = 0;
    for (int j = 0; j < B_; ++j) {
        int lj = len[j];
        rank += (lj > L0) || (lj == L0 && j < b);
    }
    ord[rank] = b;
    __syncthreads();
    int src = ord[b];
    order[b] = src;
    int dl = len[src] - 1;
    declen[b] = dl;
    out_order[b]  = (float)src;
    out_declen[b] = (float)dl;
    for (int l = 0; l < LCAP; ++l) {
        int cv = captions[src * LCAP + l];
        caps_s[b * LCAP + l]   = cv;
        out_caps[b * LCAP + l] = (float)cv;
    }
}

__global__ void bias_sum_kernel(const float* __restrict__ a,
                                const float* __restrict__ b,
                                float* __restrict__ o, int n) {
    int i = blockIdx.x * blockDim.x + threadIdx.x;
    if (i < n) o[i] = a[i] + b[i];
}

__global__ void f32_to_bf16_kernel(const float* __restrict__ s,
                                   uint16_t* __restrict__ d, int n) {
    int i = blockIdx.x * blockDim.x + threadIdx.x;
    if (i < n) d[i] = f2bf(s[i]);
}

// Gather image rows per sorted order and convert to bf16 (L2-resident copy).
__global__ void gather_img_bf16(const float* __restrict__ img,
                                const int* __restrict__ order,
                                uint16_t* __restrict__ dst, int total) {
    int id = blockIdx.x * blockDim.x + threadIdx.x;
    if (id >= total) return;
    int d  = id & (D_ - 1);
    int bp = id >> 11;
    int p  = bp % P_;
    int b  = bp / P_;
    float v = img[((size_t)order[b] * P_ + p) * D_ + d];
    dst[id] = f2bf(v);
}

// mean over pixels -> bf16 (GEMM A operand for h0/c0)
__global__ void mean_img_bf16(const float* __restrict__ img,
                              const int* __restrict__ order,
                              uint16_t* __restrict__ meanb) {
    int b = blockIdx.x;
    const float* base = img + (size_t)order[b] * P_ * D_;
    for (int d = threadIdx.x; d < D_; d += blockDim.x) {
        float s = 0.f;
        for (int p = 0; p < P_; ++p) s += base[p * D_ + d];
        meanb[b * D_ + d] = f2bf(s * (1.f / (float)P_));
    }
}

// ---------------------------------------------------------------------------
// Pack B (KxN f32, optionally concat of two sources split at `split`) into the
// CDNA5 bf16 B-fragment layout: per (ntile,kchunk) 32 lanes x 16 bf16.
// lane: n = ntile*16 + (lane&15); kbase = (lane&16)?8:0; e -> k = 32*kc+kbase+(e&7)+((e&8)?16:0)
// ---------------------------------------------------------------------------
__global__ void pack_b_bf16(const float* __restrict__ srcA,
                            const float* __restrict__ srcB, int split,
                            int K, int N, uint16_t* __restrict__ dst) {
    int id = blockIdx.x * blockDim.x + threadIdx.x;
    if (id >= N * K) return;
    int e    = id & 15;
    int lane = (id >> 4) & 31;
    int rem  = id >> 9;
    int KC   = K >> 5;
    int kc   = rem % KC;
    int nt   = rem / KC;
    int n    = nt * 16 + (lane & 15);
    int kb   = (lane & 16) ? 8 : 0;
    int k    = kc * 32 + kb + (e & 7) + ((e & 8) ? 16 : 0);
    float v  = (k < split) ? srcA[(size_t)k * N + n]
                           : srcB[(size_t)(k - split) * N + n];
    dst[id] = f2bf(v);
}

// ---------------------------------------------------------------------------
// Generic bf16 WMMA GEMM: C(MxN,f32) = A(MxK bf16 row-major) * Bp(packed) + bias
// Wave tile: 32 x (NT*16), branch-free (NT/MODE are template params).
// All NT B-fragments are loaded into distinct registers before the WMMA burst
// so loads clause together and overlap the matrix ops.
// MODE: 0 f32 store, 1 bf16 store, 2 sigmoid store,
//       3 masked vocab store (row active iff t < declen[row], else 0).
// ---------------------------------------------------------------------------
template <int NT, int MODE>
__global__ __launch_bounds__(256) void gemm_bf16_wmma(
    const uint16_t* __restrict__ A, int lda,
    const uint4* __restrict__ Bp,
    const float* __restrict__ bias,
    void* __restrict__ Cout, long long ldc,
    int M, int N, int K,
    const int* __restrict__ declen, int t) {
    const int KC     = K >> 5;
    const int tilesN = (N / 16) / NT;
    const int tilesM = M >> 5;
    int w = blockIdx.x * 8 + (threadIdx.x >> 5);
    if (w >= tilesM * tilesN) return;
    int tm   = w / tilesN;
    int tn   = w % tilesN;
    int lane = threadIdx.x & 31;

    // A fragment addressing (16-bit A 16x32 layout):
    // lanes 0-15: M=lane, K pairs {0,2,4,6}+{16,18,20,22}; lanes 16-31: +8 on K base
    int mrow = lane & 15;
    int kb   = (lane & 16) ? 8 : 0;
    int koff[8];
#pragma unroll
    for (int j = 0; j < 4; ++j) { koff[j] = kb + 2 * j; koff[j + 4] = kb + 16 + 2 * j; }
    const uint16_t* arow0 = A + (size_t)(tm * 32 + mrow) * lda;
    const uint16_t* arow1 = arow0 + (size_t)16 * lda;

    const uint4* bb[NT];
#pragma unroll
    for (int j = 0; j < NT; ++j)
        bb[j] = Bp + ((size_t)(tn * NT + j) * KC * 32 + lane) * 2;

    v8f acc[2][NT];
#pragma unroll
    for (int i = 0; i < 2; ++i)
#pragma unroll
        for (int j = 0; j < NT; ++j)
#pragma unroll
            for (int r = 0; r < 8; ++r) acc[i][j][r] = 0.f;

    for (int kc = 0; kc < KC; ++kc) {
        FragA a0, a1;
        int kbase = kc * 32;
#pragma unroll
        for (int j = 0; j < 8; ++j) {
            a0.u[j] = *(const uint32_t*)(arow0 + kbase + koff[j]);
            a1.u[j] = *(const uint32_t*)(arow1 + kbase + koff[j]);
        }
        FragB bf[NT];
#pragma unroll
        for (int j = 0; j < NT; ++j) {
            const uint4* bp = bb[j] + (size_t)kc * 64;
            bf[j].q[0] = bp[0];
            bf[j].q[1] = bp[1];
        }
#pragma unroll
        for (int j = 0; j < NT; ++j) {
            acc[0][j] = __builtin_amdgcn_wmma_f32_16x16x32_bf16(
                false, a0.v, false, bf[j].v, (short)0, acc[0][j], false, false);
            acc[1][j] = __builtin_amdgcn_wmma_f32_16x16x32_bf16(
                false, a1.v, false, bf[j].v, (short)0, acc[1][j], false, false);
        }
    }

    // Epilogue (C/D layout: VGPR r -> row r (+8 for upper lane half), col = lane&15)
    int rhalf = (lane >> 4) & 1;
#pragma unroll
    for (int i = 0; i < 2; ++i) {
#pragma unroll
        for (int j = 0; j < NT; ++j) {
            int col  = (tn * NT + j) * 16 + (lane & 15);
            float bv = bias ? bias[col] : 0.f;
#pragma unroll
            for (int r = 0; r < 8; ++r) {
                int row  = tm * 32 + i * 16 + rhalf * 8 + r;
                float v  = acc[i][j][r] + bv;
                size_t o = (size_t)row * (size_t)ldc + col;
                if constexpr (MODE == 0) {
                    ((float*)Cout)[o] = v;
                } else if constexpr (MODE == 1) {
                    ((uint16_t*)Cout)[o] = f2bf(v);
                } else if constexpr (MODE == 2) {
                    ((float*)Cout)[o] = sigm(v);
                } else {
                    ((float*)Cout)[o] = (t < declen[row]) ? v : 0.f;
                }
            }
        }
    }
}

// ---------------------------------------------------------------------------
// score[b,p] = relu(att1[b,p,:] + att2[b,:]) . full_w + full_b   (one wave/row)
// ---------------------------------------------------------------------------
__global__ __launch_bounds__(256) void score_kernel(
    const uint16_t* __restrict__ att1, const float* __restrict__ att2,
    const float* __restrict__ fullw, const float* __restrict__ fullb,
    float* __restrict__ score) {
    int w = blockIdx.x * 8 + (threadIdx.x >> 5);
    if (w >= B_ * P_) return;
    int lane = threadIdx.x & 31;
    int b = w / P_;
    const uint16_t* a1 = att1 + (size_t)w * A_;
    const float*    a2 = att2 + (size_t)b * A_;
    float s = 0.f;
#pragma unroll
    for (int i = 0; i < 16; ++i) {
        int a = lane * 16 + i;
        float v = bf2f(a1[a]) + a2[a];
        v = v > 0.f ? v : 0.f;
        s += v * fullw[a];
    }
#pragma unroll
    for (int off = 16; off; off >>= 1) s += __shfl_xor(s, off, 32);
    if (lane == 0) score[w] = s + fullb[0];
}

// ---------------------------------------------------------------------------
// Per-b softmax over 196 pixels + weighted context sum over bf16 img (L2-hot).
// ---------------------------------------------------------------------------
__global__ __launch_bounds__(256) void softmax_ctx_kernel(
    const float* __restrict__ score, const uint16_t* __restrict__ imgL,
    const int* __restrict__ declen, int t,
    float* __restrict__ alphas_out, float* __restrict__ ctx) {
    __shared__ float sa[256];
    __shared__ float red[256];
    int b = blockIdx.x, tid = threadIdx.x;
    float v = (tid < P_) ? score[b * P_ + tid] : -3.0e38f;
    red[tid] = v;
    __syncthreads();
    for (int s = 128; s; s >>= 1) {
        if (tid < s) red[tid] = fmaxf(red[tid], red[tid + s]);
        __syncthreads();
    }
    float mx = red[0];
    __syncthreads();
    float e = (tid < P_) ? __expf(v - mx) : 0.f;
    red[tid] = e;
    __syncthreads();
    for (int s = 128; s; s >>= 1) {
        if (tid < s) red[tid] += red[tid + s];
        __syncthreads();
    }
    float alpha = e / red[0];
    sa[tid] = alpha;
    __syncthreads();
    bool act = (t < declen[b]);
    if (tid < P_)
        alphas_out[((size_t)b * T_ + t) * P_ + tid] = act ? alpha : 0.f;

    float acc[8];
#pragma unroll
    for (int k = 0; k < 8; ++k) acc[k] = 0.f;
    const uint16_t* base = imgL + (size_t)b * P_ * D_ + tid * 8;
    for (int p = 0; p < P_; ++p) {
        float a = sa[p];
        uint4 q = *(const uint4*)(base + (size_t)p * D_);
        acc[0] += a * bf2f((uint16_t)(q.x & 0xffff));
        acc[1] += a * bf2f((uint16_t)(q.x >> 16));
        acc[2] += a * bf2f((uint16_t)(q.y & 0xffff));
        acc[3] += a * bf2f((uint16_t)(q.y >> 16));
        acc[4] += a * bf2f((uint16_t)(q.z & 0xffff));
        acc[5] += a * bf2f((uint16_t)(q.z >> 16));
        acc[6] += a * bf2f((uint16_t)(q.w & 0xffff));
        acc[7] += a * bf2f((uint16_t)(q.w >> 16));
    }
    float* c = ctx + (size_t)b * D_ + tid * 8;
#pragma unroll
    for (int k = 0; k < 8; ++k) c[k] = acc[k];
}

// Build bf16 LSTM input x = [emb_t, gate*ctx, h]  (128 x 3072)
__global__ void build_xh_kernel(const float* __restrict__ emb_table,
                                const int* __restrict__ caps_s, int t,
                                const float* __restrict__ gate,
                                const float* __restrict__ ctx,
                                const float* __restrict__ h,
                                uint16_t* __restrict__ xh) {
    int id = blockIdx.x * blockDim.x + threadIdx.x;
    if (id >= B_ * 3072) return;
    int b = id / 3072, k = id % 3072;
    float v;
    if (k < E_) {
        v = emb_table[(size_t)caps_s[b * LCAP + t] * E_ + k];
    } else if (k < E_ + D_) {
        int d = k - E_;
        v = gate[b * D_ + d] * ctx[b * D_ + d];
    } else {
        v = h[b * H_ + (k - E_ - D_)];
    }
    xh[id] = f2bf(v);
}

// LSTM pointwise + masking; emits f32 h/c (next step) and bf16 copies.
__global__ void lstm_kernel(const float* __restrict__ gates,
                            float* __restrict__ h, float* __restrict__ c,
                            uint16_t* __restrict__ h_bf,
                            uint16_t* __restrict__ hnew_bf,
                            const int* __restrict__ declen, int t) {
    int id = blockIdx.x * blockDim.x + threadIdx.x;
    if (id >= B_ * H_) return;
    int b = id >> 9;
    int j = id & (H_ - 1);
    const float* g = gates + (size_t)b * (4 * H_);
    float gi = sigm(g[j]);
    float gf = sigm(g[H_ + j]);
    float gg = tanhf(g[2 * H_ + j]);
    float go = sigm(g[3 * H_ + j]);
    float cn = gf * c[id] + gi * gg;
    float hn = go * tanhf(cn);
    bool act = (t < declen[b]);
    float h2 = act ? hn : h[id];
    float c2 = act ? cn : c[id];
    h[id] = h2;
    c[id] = c2;
    hnew_bf[id] = f2bf(hn);
    h_bf[id]    = f2bf(h2);
}

// ---------------------------------------------------------------------------
static inline int cdiv(long long a, long long b) { return (int)((a + b - 1) / b); }

extern "C" void kernel_launch(void* const* d_in, const int* in_sizes, int n_in,
                              void* d_out, int out_size, void* d_ws, size_t ws_size,
                              hipStream_t stream) {
    (void)in_sizes; (void)n_in; (void)out_size; (void)ws_size;
    const float* img        = (const float*)d_in[0];
    const int*   captions   = (const int*)d_in[1];
    const int*   lengths    = (const int*)d_in[2];
    const float* enc_att_w  = (const float*)d_in[3];
    const float* enc_att_b  = (const float*)d_in[4];
    const float* dec_att_w  = (const float*)d_in[5];
    const float* dec_att_b  = (const float*)d_in[6];
    const float* full_att_w = (const float*)d_in[7];
    const float* full_att_b = (const float*)d_in[8];
    const float* emb_table  = (const float*)d_in[9];
    const float* w_ih       = (const float*)d_in[10];
    const float* w_hh       = (const float*)d_in[11];
    const float* b_ih       = (const float*)d_in[12];
    const float* b_hh       = (const float*)d_in[13];
    const float* init_h_w   = (const float*)d_in[14];
    const float* init_h_b   = (const float*)d_in[15];
    const float* init_c_w   = (const float*)d_in[16];
    const float* init_c_b   = (const float*)d_in[17];
    const float* f_beta_w   = (const float*)d_in[18];
    const float* f_beta_b   = (const float*)d_in[19];
    const float* fc_w       = (const float*)d_in[20];
    const float* fc_b       = (const float*)d_in[21];

    // Output layout (flat, return order)
    float* out      = (float*)d_out;
    float* preds_o  = out;                                 // 128*31*10000
    float* caps_o   = preds_o + (size_t)B_ * T_ * V_;      // 128*32
    float* declen_o = caps_o + (size_t)B_ * LCAP;          // 128
    float* alphas_o = declen_o + B_;                       // 128*31*196
    float* order_o  = alphas_o + (size_t)B_ * T_ * P_;     // 128

    // Workspace bump allocator
    size_t off = 0;
    char* wsb = (char*)d_ws;
    auto alloc = [&](size_t bytes) -> char* {
        char* p = wsb + off;
        off += (bytes + 255) & ~(size_t)255;
        return p;
    };
    uint16_t* imgL    = (uint16_t*)alloc((size_t)B_ * P_ * D_ * 2);
    uint16_t* att1    = (uint16_t*)alloc((size_t)B_ * P_ * A_ * 2);
    uint16_t* encWp   = (uint16_t*)alloc((size_t)D_ * A_ * 2);
    uint16_t* decWp   = (uint16_t*)alloc((size_t)H_ * A_ * 2);
    uint16_t* fbetaWp = (uint16_t*)alloc((size_t)H_ * D_ * 2);
    uint16_t* wcatP   = (uint16_t*)alloc((size_t)3072 * (4 * H_) * 2);
    uint16_t* fcWp    = (uint16_t*)alloc((size_t)H_ * V_ * 2);
    uint16_t* ihWp    = (uint16_t*)alloc((size_t)D_ * H_ * 2);
    uint16_t* icWp    = (uint16_t*)alloc((size_t)D_ * H_ * 2);
    uint16_t* meanb   = (uint16_t*)alloc((size_t)B_ * D_ * 2);
    float*    hbuf    = (float*)alloc((size_t)B_ * H_ * 4);
    float*    cbuf    = (float*)alloc((size_t)B_ * H_ * 4);
    uint16_t* h_bf    = (uint16_t*)alloc((size_t)B_ * H_ * 2);
    uint16_t* hnew_bf = (uint16_t*)alloc((size_t)B_ * H_ * 2);
    float*    att2    = (float*)alloc((size_t)B_ * A_ * 4);
    float*    gate    = (float*)alloc((size_t)B_ * D_ * 4);
    float*    score   = (float*)alloc((size_t)B_ * P_ * 4);
    float*    ctx     = (float*)alloc((size_t)B_ * D_ * 4);
    uint16_t* xh      = (uint16_t*)alloc((size_t)B_ * 3072 * 2);
    float*    gatesb  = (float*)alloc((size_t)B_ * 4 * H_ * 4);
    int*      order   = (int*)alloc(B_ * 4);
    int*      declen  = (int*)alloc(B_ * 4);
    int*      caps_s  = (int*)alloc(B_ * LCAP * 4);
    float*    bsum    = (float*)alloc(4 * H_ * 4);

    // ---- Setup phase -------------------------------------------------------
    sort_kernel<<<1, B_, 0, stream>>>(lengths, captions, order, declen, caps_s,
                                      caps_o, declen_o, order_o);
    bias_sum_kernel<<<cdiv(4 * H_, 256), 256, 0, stream>>>(b_ih, b_hh, bsum, 4 * H_);
    {
        int total = B_ * P_ * D_;
        gather_img_bf16<<<cdiv(total, 256), 256, 0, stream>>>(img, order, imgL, total);
    }
    mean_img_bf16<<<B_, 256, 0, stream>>>(img, order, meanb);

    // Pack static weights into B-fragment layout
    pack_b_bf16<<<cdiv((long long)D_ * A_, 256), 256, 0, stream>>>(enc_att_w, enc_att_w, D_, D_, A_, encWp);
    pack_b_bf16<<<cdiv((long long)H_ * A_, 256), 256, 0, stream>>>(dec_att_w, dec_att_w, H_, H_, A_, decWp);
    pack_b_bf16<<<cdiv((long long)H_ * D_, 256), 256, 0, stream>>>(f_beta_w, f_beta_w, H_, H_, D_, fbetaWp);
    pack_b_bf16<<<cdiv((long long)3072 * 4 * H_, 256), 256, 0, stream>>>(w_ih, w_hh, E_ + D_, 3072, 4 * H_, wcatP);
    pack_b_bf16<<<cdiv((long long)H_ * V_, 256), 256, 0, stream>>>(fc_w, fc_w, H_, H_, V_, fcWp);
    pack_b_bf16<<<cdiv((long long)D_ * H_, 256), 256, 0, stream>>>(init_h_w, init_h_w, D_, D_, H_, ihWp);
    pack_b_bf16<<<cdiv((long long)D_ * H_, 256), 256, 0, stream>>>(init_c_w, init_c_w, D_, D_, H_, icWp);

    // h0 / c0 : (128x2048)(2048x512), 32x64 wave tiles
    {
        int waves = (B_ / 32) * (H_ / 64);
        gemm_bf16_wmma<4, 0><<<cdiv(waves, 8), 256, 0, stream>>>(
            meanb, D_, (const uint4*)ihWp, init_h_b, hbuf, H_, B_, H_, D_, nullptr, 0);
        gemm_bf16_wmma<4, 0><<<cdiv(waves, 8), 256, 0, stream>>>(
            meanb, D_, (const uint4*)icWp, init_c_b, cbuf, H_, B_, H_, D_, nullptr, 0);
    }
    f32_to_bf16_kernel<<<cdiv(B_ * H_, 256), 256, 0, stream>>>(hbuf, h_bf, B_ * H_);

    // att1 = img @ enc_att_w + enc_att_b  (25088x2048 x 2048x512) -> bf16
    {
        int waves = ((B_ * P_) / 32) * (A_ / 64);
        gemm_bf16_wmma<4, 1><<<cdiv(waves, 8), 256, 0, stream>>>(
            imgL, D_, (const uint4*)encWp, enc_att_b, att1, A_, B_ * P_, A_, D_, nullptr, 0);
    }

    // ---- Recurrent scan ----------------------------------------------------
    for (int t = 0; t < T_; ++t) {
        // att2 = h @ dec_att_w + b
        gemm_bf16_wmma<4, 0><<<cdiv((B_ / 32) * (A_ / 64), 8), 256, 0, stream>>>(
            h_bf, H_, (const uint4*)decWp, dec_att_b, att2, A_, B_, A_, H_, nullptr, 0);
        // gate = sigmoid(h @ f_beta_w + b)
        gemm_bf16_wmma<4, 2><<<cdiv((B_ / 32) * (D_ / 64), 8), 256, 0, stream>>>(
            h_bf, H_, (const uint4*)fbetaWp, f_beta_b, gate, D_, B_, D_, H_, nullptr, 0);
        // score
        score_kernel<<<cdiv(B_ * P_, 8), 256, 0, stream>>>(att1, att2, full_att_w, full_att_b, score);
        // softmax + context (+ alphas output)
        softmax_ctx_kernel<<<B_, 256, 0, stream>>>(score, imgL, declen, t, alphas_o, ctx);
        // x = [emb_t, gate*ctx, h] in bf16
        build_xh_kernel<<<cdiv(B_ * 3072, 256), 256, 0, stream>>>(
            emb_table, caps_s, t, gate, ctx, hbuf, xh);
        // gates = x @ [w_ih; w_hh] + (b_ih + b_hh)
        gemm_bf16_wmma<4, 0><<<cdiv((B_ / 32) * ((4 * H_) / 64), 8), 256, 0, stream>>>(
            xh, 3072, (const uint4*)wcatP, bsum, gatesb, 4 * H_, B_, 4 * H_, 3072, nullptr, 0);
        // LSTM pointwise + masking
        lstm_kernel<<<cdiv(B_ * H_, 256), 256, 0, stream>>>(
            gatesb, hbuf, cbuf, h_bf, hnew_bf, declen, t);
        // preds[:, t, :] = mask ? h_new @ fc_w + fc_b : 0   (NT=5: 32x80 tile, 10000/80=125)
        {
            int waves = (B_ / 32) * (V_ / 80);
            gemm_bf16_wmma<5, 3><<<cdiv(waves, 8), 256, 0, stream>>>(
                hnew_bf, H_, (const uint4*)fcWp, fc_b, preds_o + (size_t)t * V_,
                (long long)T_ * V_, B_, V_, H_, declen, t);
        }
    }
}
